// Encoder_32693291057793
// MI455X (gfx1250) — matrix operand order repaired and checked by reading, not compiled, not run on hardware
//
#include <hip/hip_runtime.h>
#include <hip/hip_bf16.h>

typedef __bf16 bf16;
typedef __attribute__((ext_vector_type(16))) __bf16 v16bf;
typedef __attribute__((ext_vector_type(8)))  float   v8f;

#define Bc 2
#define Sc 1024
#define Dc 1024
#define Hc 16
#define DQc 64
#define DHc 4096
#define Lc 6

__device__ inline bf16 f2bf(float x) { return (bf16)x; }

union FragU { v16bf v; unsigned int u[8]; };

// ---- CDNA5 async global->LDS staging (ASYNCcnt) -----------------------------
__device__ inline void async_b128(unsigned lds_off, const void* gaddr) {
  asm volatile("global_load_async_to_lds_b128 %0, %1, off"
               :: "v"(lds_off), "v"(gaddr) : "memory");
}
__device__ inline void wait_async() {
  asm volatile("s_wait_asynccnt 0x0" ::: "memory");
}
__device__ inline unsigned lds_addr(const void* p) {
  return (unsigned)(size_t)p;   // low 32 bits of generic LDS pointer = LDS offset
}

// A-matrix 16x32 bf16 fragment from row-major LDS tile (leading dim ld, elements).
// lanes 0-15 -> M=lane, K in {0..7,16..23}; lanes 16-31 -> M=lane-16, K in {8..15,24..31}
__device__ inline v16bf load_a_frag(const bf16* base, int ld, int lane) {
  int hi  = lane >> 4;
  int row = lane & 15;
  const bf16* r = base + row * ld;
  FragU f;
#pragma unroll
  for (int i = 0; i < 8; ++i) {
    int k = ((i & 4) ? 16 : 0) + (hi ? 8 : 0) + ((i & 3) << 1);
    f.u[i] = *(const unsigned int*)(r + k);
  }
  return f.v;
}

// B-matrix 32x16 bf16 fragment; LDS tile stored TRANSPOSED as [N][Kc] row-major.
// lanes 0-15 -> N=lane, K in {0..7,16..23}; lanes 16-31 -> N=lane-16, K in {8..15,24..31}
__device__ inline v16bf load_b_frag(const bf16* base, int ld, int lane) {
  int hi  = lane >> 4;
  int col = lane & 15;
  const bf16* r = base + col * ld;
  FragU f;
#pragma unroll
  for (int i = 0; i < 8; ++i) f.u[i] = *(const unsigned int*)(r + ((i & 4) ? 16 : 0) + (hi ? 8 : 0) + ((i & 3) << 1));
  return f.v;
}

__device__ inline v8f wmma_bf16(v16bf a, v16bf b, v8f c) {
  return __builtin_amdgcn_wmma_f32_16x16x32_bf16(false, a, false, b, (short)0, c, false, false);
}

// ---------------------------------------------------------------------------
// Double-buffered bf16 WMMA GEMM core. out = A(bf16 [M,K]) * Wt^T + bias
// Wt is bf16 [N,K] (pre-transposed). Tile 128x128x64 per 256-thread block:
// 8 waves in 4x2 grid, wave tile 32x64 -> 16 WMMAs/K-step, B-frags reused 2x.
// OUTBT: write bf16 transposed per-head (V path): Vt[B,H,DQ,S].
// ---------------------------------------------------------------------------
template<bool RELU, bool RESID, bool OUTF, bool OUTB, bool OUTBT>
__device__ void gemm_core(const bf16* __restrict__ A, const bf16* __restrict__ Wt,
                          const float* __restrict__ bias, const float* __restrict__ resid,
                          float* __restrict__ outf, bf16* __restrict__ outb,
                          bf16* __restrict__ outbt,
                          int M, int N, int K, int bm, int bn) {
  __shared__ alignas(16) bf16 sA[2][128 * 64];   // [row][k]
  __shared__ alignas(16) bf16 sB[2][128 * 64];   // [n][k]
  const int tid  = threadIdx.x;
  const int wave = tid >> 5, lane = tid & 31;
  const int wr = wave >> 1;      // 0..3: row group of 32
  const int wc = wave & 1;       // 0..1: col group of 64

  v8f acc[2][4] = {};

  auto stage = [&](int buf, int k0) {
    unsigned aB = lds_addr(&sA[buf][0]);
    unsigned bB = lds_addr(&sB[buf][0]);
#pragma unroll
    for (int i = 0; i < 4; ++i) {              // 1024 chunks of 16B (128x64)
      int e = tid + i * 256;
      int r = e >> 3, c = (e & 7) << 3;
      async_b128(aB + (unsigned)(r * 64 + c) * 2, A + (size_t)(bm + r) * K + k0 + c);
    }
#pragma unroll
    for (int i = 0; i < 4; ++i) {              // 1024 chunks of 16B (128x64)
      int e = tid + i * 256;
      int r = e >> 3, c = (e & 7) << 3;
      async_b128(bB + (unsigned)(r * 64 + c) * 2, Wt + (size_t)(bn + r) * K + k0 + c);
    }
  };

  const int nK = K >> 6;
  stage(0, 0);
  wait_async();
  __syncthreads();

  for (int s = 0; s < nK; ++s) {
    if (s + 1 < nK) stage((s + 1) & 1, (s + 1) << 6);
    const bf16* Ab = sA[s & 1] + wr * 32 * 64;
    const bf16* Bb = sB[s & 1] + wc * 64 * 64;
#pragma unroll
    for (int kk = 0; kk < 2; ++kk) {           // two K=32 chunks
      v16bf a0 = load_a_frag(Ab + 0 * 16 * 64 + kk * 32, 64, lane);
      v16bf a1 = load_a_frag(Ab + 1 * 16 * 64 + kk * 32, 64, lane);
#pragma unroll
      for (int t = 0; t < 4; ++t) {
        v16bf bfrg = load_b_frag(Bb + t * 16 * 64 + kk * 32, 64, lane);
        acc[0][t] = wmma_bf16(a0, bfrg, acc[0][t]);
        acc[1][t] = wmma_bf16(a1, bfrg, acc[1][t]);
      }
    }
    wait_async();
    __syncthreads();
  }

  const int hi8 = (lane >> 4) * 8;
#pragma unroll
  for (int rstrip = 0; rstrip < 2; ++rstrip) {
#pragma unroll
    for (int t = 0; t < 4; ++t) {
      int n = bn + wc * 64 + t * 16 + (lane & 15);
      float bv = bias[n];
#pragma unroll
      for (int v = 0; v < 8; ++v) {
        int m = bm + wr * 32 + rstrip * 16 + v + hi8;
        float val = acc[rstrip][t][v] + bv;
        if (RELU)  val = fmaxf(val, 0.0f);
        if (RESID) val += resid[(size_t)m * N + n];
        if (OUTF)  outf[(size_t)m * N + n] = val;
        if (OUTB)  outb[(size_t)m * N + n] = f2bf(val);
        if (OUTBT) // V: [B,H,DQ,S]: batch = m>>10, s = m&1023
          outbt[((size_t)(m >> 10) * Dc + n) * Sc + (m & (Sc - 1))] = f2bf(val);
      }
    }
  }
}

template<bool RELU, bool RESID, bool OUTF, bool OUTB>
__global__ __launch_bounds__(256)
void gemm_kernel(const bf16* __restrict__ A, const bf16* __restrict__ Wt,
                 const float* __restrict__ bias, const float* __restrict__ resid,
                 float* __restrict__ outf, bf16* __restrict__ outb, int M, int N, int K) {
  gemm_core<RELU, RESID, OUTF, OUTB, false>(A, Wt, bias, resid, outf, outb, nullptr,
                                            M, N, K, blockIdx.x * 128, blockIdx.y * 128);
}

// fused Q/K/V projection: blockIdx.z selects projection; V written transposed.
__global__ __launch_bounds__(256)
void qkv_kernel(const bf16* __restrict__ Xb,
                const bf16* WqT, const bf16* WkT, const bf16* WvT,
                const float* bq, const float* bk, const float* bv,
                bf16* Q, bf16* Ko, bf16* Vt, int M, int N, int K) {
  const int bm = blockIdx.x * 128, bn = blockIdx.y * 128;
  if (blockIdx.z == 2) {
    gemm_core<false, false, false, false, true>(Xb, WvT, bv, nullptr, nullptr, nullptr, Vt,
                                                M, N, K, bm, bn);
  } else {
    const bf16* Wt = (blockIdx.z == 0) ? WqT : WkT;
    const float* b = (blockIdx.z == 0) ? bq  : bk;
    bf16* out      = (blockIdx.z == 0) ? Q   : Ko;
    gemm_core<false, false, false, true, false>(Xb, Wt, b, nullptr, nullptr, out, nullptr,
                                                M, N, K, bm, bn);
  }
}

// ---------------------------------------------------------------------------
// Flash attention: grid (B*H, S/64), 128 threads (4 waves); k-tiles of 64;
// online softmax. Q/K bf16 [B*S,H*DQ]; Vt bf16 [B,H,DQ,S]; ctx bf16 [B*S,H*DQ].
// ---------------------------------------------------------------------------
__global__ __launch_bounds__(128)
void attention_kernel(const bf16* __restrict__ Q, const bf16* __restrict__ Km,
                      const bf16* __restrict__ Vt, const unsigned char* __restrict__ mask,
                      bf16* __restrict__ ctx, float scale) {
  const int bh = blockIdx.x;
  const int b = bh >> 4, h = bh & 15;
  const int q0 = blockIdx.y * 64;
  const int tid = threadIdx.x, wave = tid >> 5, lane = tid & 31;
  const int hi8 = (lane >> 4) * 8;
  const size_t rs = Hc * DQc;   // 1024

  __shared__ alignas(16) bf16 sQ[64 * 64];      // [qrow][dq]
  __shared__ alignas(16) bf16 sK[64 * 64];      // [kpos][dq]
  __shared__ alignas(16) bf16 sV[64 * 64];      // [dq][kpos]  (from Vt)
  __shared__ alignas(16) bf16 sP[4][16 * 64];   // per-wave P tile (A-layout 16x64)

  const bf16* Qb  = Q  + ((size_t)b * Sc + q0) * rs + h * DQc;
  const bf16* Kb  = Km + ((size_t)b * Sc) * rs + h * DQc;
  const bf16* Vtb = Vt + (size_t)(b * Hc + h) * DQc * Sc;
  const unsigned char* mb = mask + (size_t)b * Sc * Sc;

  {
    unsigned qB = lds_addr(&sQ[0]);
#pragma unroll
    for (int i = 0; i < 4; ++i) {            // 512 chunks of 16B (64x64)
      int e = tid + i * 128;
      int r = e >> 3, c = (e & 7) << 3;
      async_b128(qB + (unsigned)(r * 64 + c) * 2, Qb + (size_t)r * rs + c);
    }
  }
  wait_async();
  __syncthreads();

  v16bf aq0 = load_a_frag(sQ + wave * 16 * 64 + 0,  64, lane);
  v16bf aq1 = load_a_frag(sQ + wave * 16 * 64 + 32, 64, lane);

  v8f acc[4] = {};
  float m_i[8], l_i[8];
#pragma unroll
  for (int v = 0; v < 8; ++v) { m_i[v] = -1e30f; l_i[v] = 0.0f; }

  for (int k0 = 0; k0 < Sc; k0 += 64) {
    unsigned kB = lds_addr(&sK[0]);
    unsigned vB = lds_addr(&sV[0]);
#pragma unroll
    for (int i = 0; i < 4; ++i) {            // K tile: 512 chunks (64x64)
      int e = tid + i * 128;
      int r = e >> 3, c = (e & 7) << 3;
      async_b128(kB + (unsigned)(r * 64 + c) * 2, Kb + (size_t)(k0 + r) * rs + c);
    }
#pragma unroll
    for (int i = 0; i < 4; ++i) {            // V tile from Vt: rows=dq, cols=kpos
      int e = tid + i * 128;
      int r = e >> 3, c = (e & 7) << 3;
      async_b128(vB + (unsigned)(r * 64 + c) * 2, Vtb + (size_t)r * Sc + k0 + c);
    }
    wait_async();
    __syncthreads();

    // scores: 16x64 strip = four 16x16 C tiles, contraction over DQ=64
    v8f sc[4] = {};
#pragma unroll
    for (int t = 0; t < 4; ++t) {
      sc[t] = wmma_bf16(aq0, load_b_frag(sK + t * 16 * 64 + 0,  64, lane), sc[t]);
      sc[t] = wmma_bf16(aq1, load_b_frag(sK + t * 16 * 64 + 32, 64, lane), sc[t]);
    }

    float alpha_r[8];
#pragma unroll
    for (int v = 0; v < 8; ++v) {
      int qq = q0 + wave * 16 + v + hi8;
      const unsigned char* mr = mb + (size_t)qq * Sc + k0 + (lane & 15);
      float x[4];
#pragma unroll
      for (int t = 0; t < 4; ++t)
        x[t] = mr[t * 16] ? -1e9f : sc[t][v] * scale;

      float rm = fmaxf(fmaxf(x[0], x[1]), fmaxf(x[2], x[3]));
#pragma unroll
      for (int o = 1; o < 16; o <<= 1) rm = fmaxf(rm, __shfl_xor(rm, o, 32));
      float mx = fmaxf(m_i[v], rm);
      alpha_r[v] = __expf(m_i[v] - mx);
      m_i[v] = mx;
      float p[4], rsum = 0.0f;
#pragma unroll
      for (int t = 0; t < 4; ++t) { p[t] = __expf(x[t] - mx); rsum += p[t]; }
#pragma unroll
      for (int o = 1; o < 16; o <<= 1) rsum += __shfl_xor(rsum, o, 32);
      l_i[v] = l_i[v] * alpha_r[v] + rsum;

      int mrow = v + hi8;
#pragma unroll
      for (int t = 0; t < 4; ++t)
        sP[wave][mrow * 64 + t * 16 + (lane & 15)] = f2bf(p[t]);
    }
#pragma unroll
    for (int t = 0; t < 4; ++t)
#pragma unroll
      for (int v = 0; v < 8; ++v) acc[t][v] *= alpha_r[v];
    __syncthreads();   // sP visible (cross-lane within wave)

    v16bf ap0 = load_a_frag(&sP[wave][0],  64, lane);   // K = kpos 0..31
    v16bf ap1 = load_a_frag(&sP[wave][32], 64, lane);   // K = kpos 32..63
#pragma unroll
    for (int t = 0; t < 4; ++t) {
      acc[t] = wmma_bf16(ap0, load_b_frag(sV + t * 16 * 64 + 0,  64, lane), acc[t]);
      acc[t] = wmma_bf16(ap1, load_b_frag(sV + t * 16 * 64 + 32, 64, lane), acc[t]);
    }
    __syncthreads();   // done reading sK/sV before next restage
  }

  bf16* cb = ctx + ((size_t)b * Sc) * rs + h * DQc;
#pragma unroll
  for (int t = 0; t < 4; ++t)
#pragma unroll
    for (int v = 0; v < 8; ++v) {
      int m = q0 + wave * 16 + v + hi8;
      int d = t * 16 + (lane & 15);
      cb[(size_t)m * rs + d] = f2bf(acc[t][v] / l_i[v]);
    }
}

// ---------------------------------------------------------------------------
// LayerNorm over D=1024; one block per row; writes f32 state + bf16 copy.
// ---------------------------------------------------------------------------
__global__ __launch_bounds__(256)
void layernorm_kernel(const float* __restrict__ Y, const float* __restrict__ g,
                      const float* __restrict__ be, float* __restrict__ X,
                      bf16* __restrict__ Xb) {
  const int row = blockIdx.x, tid = threadIdx.x;
  const float* y = Y + (size_t)row * Dc;
  __shared__ float red[8];
  float v[4]; float s = 0.0f;
#pragma unroll
  for (int i = 0; i < 4; ++i) { v[i] = y[tid + i * 256]; s += v[i]; }
#pragma unroll
  for (int o = 16; o > 0; o >>= 1) s += __shfl_xor(s, o, 32);
  if ((tid & 31) == 0) red[tid >> 5] = s;
  __syncthreads();
  float mean = 0.0f;
#pragma unroll
  for (int i = 0; i < 8; ++i) mean += red[i];
  mean *= (1.0f / Dc);
  __syncthreads();
  float s2 = 0.0f;
#pragma unroll
  for (int i = 0; i < 4; ++i) { float d = v[i] - mean; s2 += d * d; }
#pragma unroll
  for (int o = 16; o > 0; o >>= 1) s2 += __shfl_xor(s2, o, 32);
  if ((tid & 31) == 0) red[tid >> 5] = s2;
  __syncthreads();
  float var = 0.0f;
#pragma unroll
  for (int i = 0; i < 8; ++i) var += red[i];
  var *= (1.0f / Dc);
  float rstd = rsqrtf(var + 1e-5f);
#pragma unroll
  for (int i = 0; i < 4; ++i) {
    int c = tid + i * 256;
    float o = (v[i] - mean) * rstd * g[c] + be[c];
    X[(size_t)row * Dc + c]  = o;
    Xb[(size_t)row * Dc + c] = f2bf(o);
  }
}

__global__ void convert_kernel(const float* __restrict__ in, float* __restrict__ X,
                               bf16* __restrict__ Xb, int n) {
  int i = blockIdx.x * 256 + threadIdx.x;
  if (i < n) { float v = in[i]; X[i] = v; Xb[i] = f2bf(v); }
}

// Tiled transpose + f32->bf16: W [K,N] -> Wt [N,K]
__global__ __launch_bounds__(256)
void wtranspose_kernel(const float* __restrict__ W, bf16* __restrict__ Wt, int K, int N) {
  __shared__ bf16 t[64][65];
  const int bk = blockIdx.x * 64, bn = blockIdx.y * 64, tid = threadIdx.x;
  for (int i = tid; i < 64 * 64; i += 256) {
    int r = i >> 6, c = i & 63;
    t[c][r] = f2bf(W[(size_t)(bk + r) * N + bn + c]);
  }
  __syncthreads();
  for (int i = tid; i < 64 * 64; i += 256) {
    int r = i >> 6, c = i & 63;
    Wt[(size_t)(bn + r) * K + bk + c] = t[r][c];
  }
}

extern "C" void kernel_launch(void* const* d_in, const int* in_sizes, int n_in,
                              void* d_out, int out_size, void* d_ws, size_t ws_size,
                              hipStream_t stream) {
  (void)in_sizes; (void)n_in; (void)out_size; (void)ws_size;
  const int M = Bc * Sc;  // 2048

  const float* x    = (const float*)d_in[0];
  const unsigned char* mask = (const unsigned char*)d_in[1];
  const float* Wq = (const float*)d_in[2];  const float* bq = (const float*)d_in[3];
  const float* Wk = (const float*)d_in[4];  const float* bk = (const float*)d_in[5];
  const float* Wv = (const float*)d_in[6];  const float* bv = (const float*)d_in[7];
  const float* Wo = (const float*)d_in[8];  const float* bo = (const float*)d_in[9];
  const float* g1 = (const float*)d_in[10]; const float* be1 = (const float*)d_in[11];
  const float* W1 = (const float*)d_in[12]; const float* b1 = (const float*)d_in[13];
  const float* W2 = (const float*)d_in[14]; const float* b2 = (const float*)d_in[15];
  const float* g2 = (const float*)d_in[16]; const float* be2 = (const float*)d_in[17];

  char* ws = (char*)d_ws;
  size_t off = 0;
  auto alloc = [&](size_t bytes) { void* p = ws + off; off += (bytes + 255) & ~(size_t)255; return p; };
  float* X  = (float*)alloc((size_t)M * Dc * 4);
  bf16*  Xb = (bf16*) alloc((size_t)M * Dc * 2);
  float* Yf = (float*)alloc((size_t)M * Dc * 4);
  bf16*  Qb = (bf16*) alloc((size_t)M * Dc * 2);
  bf16*  Kb = (bf16*) alloc((size_t)M * Dc * 2);
  bf16*  Vt = (bf16*) alloc((size_t)M * Dc * 2);   // [B,H,DQ,S]
  bf16*  Cb = (bf16*) alloc((size_t)M * Dc * 2);
  bf16*  Fb = (bf16*) alloc((size_t)M * DHc * 2);
  bf16*  WqT = (bf16*)alloc((size_t)Lc * Dc * Dc * 2);
  bf16*  WkT = (bf16*)alloc((size_t)Lc * Dc * Dc * 2);
  bf16*  WvT = (bf16*)alloc((size_t)Lc * Dc * Dc * 2);
  bf16*  WoT = (bf16*)alloc((size_t)Lc * Dc * Dc * 2);
  bf16*  W1T = (bf16*)alloc((size_t)Lc * Dc * DHc * 2);
  bf16*  W2T = (bf16*)alloc((size_t)Lc * DHc * Dc * 2);

  for (int l = 0; l < Lc; ++l) {
    size_t od = (size_t)l * Dc * Dc, oh = (size_t)l * Dc * DHc;
    wtranspose_kernel<<<dim3(16, 16), 256, 0, stream>>>(Wq + od, WqT + od, Dc, Dc);
    wtranspose_kernel<<<dim3(16, 16), 256, 0, stream>>>(Wk + od, WkT + od, Dc, Dc);
    wtranspose_kernel<<<dim3(16, 16), 256, 0, stream>>>(Wv + od, WvT + od, Dc, Dc);
    wtranspose_kernel<<<dim3(16, 16), 256, 0, stream>>>(Wo + od, WoT + od, Dc, Dc);
    wtranspose_kernel<<<dim3(16, 64), 256, 0, stream>>>(W1 + oh, W1T + oh, Dc, DHc);
    wtranspose_kernel<<<dim3(64, 16), 256, 0, stream>>>(W2 + oh, W2T + oh, DHc, Dc);
  }

  convert_kernel<<<(M * Dc + 255) / 256, 256, 0, stream>>>(x, X, Xb, M * Dc);

  const dim3 gQKV(M / 128, Dc / 128, 3);
  const dim3 gD(M / 128, Dc / 128);
  const dim3 gH(M / 128, DHc / 128);
  const dim3 gA(Bc * Hc, Sc / 64);

  for (int l = 0; l < Lc; ++l) {
    size_t od = (size_t)l * Dc * Dc, oh = (size_t)l * Dc * DHc;
    const float* bql = bq + (size_t)l * Dc;  const float* bkl = bk + (size_t)l * Dc;
    const float* bvl = bv + (size_t)l * Dc;  const float* bol = bo + (size_t)l * Dc;
    const float* b1l = b1 + (size_t)l * DHc; const float* b2l = b2 + (size_t)l * Dc;
    const float* g1l = g1 + (size_t)l * Dc;  const float* be1l = be1 + (size_t)l * Dc;
    const float* g2l = g2 + (size_t)l * Dc;  const float* be2l = be2 + (size_t)l * Dc;

    qkv_kernel<<<gQKV, 256, 0, stream>>>(Xb, WqT + od, WkT + od, WvT + od,
                                         bql, bkl, bvl, Qb, Kb, Vt, M, Dc, Dc);

    attention_kernel<<<gA, 128, 0, stream>>>(Qb, Kb, Vt, mask, Cb, 0.125f);

    gemm_kernel<false, true, true, false><<<gD, 256, 0, stream>>>(
        Cb, WoT + od, bol, X, Yf, nullptr, M, Dc, Dc);
    layernorm_kernel<<<M, 256, 0, stream>>>(Yf, g1l, be1l, X, Xb);

    gemm_kernel<true, false, false, true><<<gH, 256, 0, stream>>>(
        Xb, W1T + oh, b1l, nullptr, nullptr, Fb, M, DHc, Dc);
    gemm_kernel<false, true, true, false><<<gD, 256, 0, stream>>>(
        Fb, W2T + oh, b2l, X, Yf, nullptr, M, Dc, DHc);
    layernorm_kernel<<<M, 256, 0, stream>>>(Yf, g2l, be2l, X, Xb);
  }

  hipMemcpyAsync(d_out, X, (size_t)M * Dc * sizeof(float),
                 hipMemcpyDeviceToDevice, stream);
}
